// CMLCell_17102559773381
// MI455X (gfx1250) — matrix operand order repaired and checked by reading, not compiled, not run on hardware
//
#include <hip/hip_runtime.h>
#include <hip/hip_bf16.h>
#include <hip/hip_fp16.h>

typedef __attribute__((ext_vector_type(16))) _Float16 v16h;
typedef __attribute__((ext_vector_type(8)))  _Float16 v8h;
typedef __attribute__((ext_vector_type(8)))  float    v8f;

constexpr int kVocab    = 50257;
constexpr int kVocabPad = 50264;           // multiple of 8 -> 16B-aligned f16 rows
constexpr int kD     = 384;
constexpr int kDS    = 512;
constexpr int kH     = 4;
constexpr int kM     = 32;
constexpr int kDH    = 96;                 // kD / kH
constexpr int kRank  = 32;
constexpr int kB     = 8;
constexpr int kT     = 512;
constexpr int kBT    = kB * kT;            // 4096
constexpr int kSreal = 1 + kM + kT;        // 545
constexpr int kSP    = 576;                // padded to multiple of 32

// load a 4-byte f16 pair into fragment slots [2i, 2i+1]
__device__ __forceinline__ void ld_pair(const _Float16* p, v16h& f, int i) {
  union { unsigned u; _Float16 h[2]; } t;
  t.u = *(const unsigned*)p;
  f[2 * i] = t.h[0]; f[2 * i + 1] = t.h[1];
}

// ---------------------------------------------------------------------------
// f32 -> f16 cast (grid-stride)
// ---------------------------------------------------------------------------
__global__ void k_cast_f16(const float* __restrict__ src, _Float16* __restrict__ dst, long n) {
  long i = (long)blockIdx.x * blockDim.x + threadIdx.x;
  long stride = (long)gridDim.x * blockDim.x;
  for (; i < n; i += stride) dst[i] = (_Float16)src[i];
}

// f32 (kD x kVocab) -> f16 (kD x kVocabPad), zero-filled pad columns
__global__ void k_cast_pad_f16(const float* __restrict__ src, _Float16* __restrict__ dst) {
  long n = (long)kD * kVocabPad;
  long i = (long)blockIdx.x * blockDim.x + threadIdx.x;
  long stride = (long)gridDim.x * blockDim.x;
  for (; i < n; i += stride) {
    int r = (int)(i / kVocabPad), c = (int)(i % kVocabPad);
    dst[i] = (c < kVocab) ? (_Float16)src[(long)r * kVocab + c] : (_Float16)0.f;
  }
}

// ---------------------------------------------------------------------------
// embedding gather -> f16
// ---------------------------------------------------------------------------
__global__ void k_gather_emb(const int* __restrict__ tok, const float* __restrict__ emb,
                             _Float16* __restrict__ xe) {
  long i = (long)blockIdx.x * blockDim.x + threadIdx.x;
  long n = (long)kBT * kD;
  long stride = (long)gridDim.x * blockDim.x;
  for (; i < n; i += stride) {
    long row = i / kD;
    int  col = (int)(i % kD);
    xe[i] = (_Float16)emb[(long)tok[row] * kD + col];
  }
}

// ---------------------------------------------------------------------------
// Generic WMMA GEMM: C = A(MxK) * B(KxN) (+bias), f16 in, f32 accumulate.
// Block tile 128x64, 8 waves, wave tile 32x32, K step 32.
// Software-pipelined, double-buffered LDS, one barrier per K-step.
// A tile staged with global_load_async_to_lds_b128 (ASYNCcnt).
// Preconditions: K%32==0, lda%8==0, ldb%8==0, 16B-aligned pointers,
//                B rows readable to ldb boundary (pad provides slack).
// ---------------------------------------------------------------------------
__global__ __launch_bounds__(256) void k_gemm_f16(
    const _Float16* __restrict__ A, int lda,
    const _Float16* __restrict__ B, int ldb,
    const float* __restrict__ bias,
    float* __restrict__ Cf, int ldcf,
    _Float16* __restrict__ Ch, int ldch,
    int M, int N, int K)
{
  __shared__ _Float16 As[2][128][40];  // row stride 80B (16B aligned)
  __shared__ _Float16 Bs[2][64][40];   // transposed: [n][k]

  const int tid  = threadIdx.x;
  const int wave = tid >> 5;
  const int lane = tid & 31;
  const int hl   = lane >> 4;
  const int l16  = lane & 15;
  const int bm   = blockIdx.y * 128;
  const int bn   = blockIdx.x * 64;
  const int wm   = (wave >> 1) * 32;
  const int wn   = (wave & 1) * 32;
  const bool fullM = (bm + 128 <= M);      // block-uniform

  const int bkk = tid >> 3;                // 0..31  (B staging k)
  const int bnn = (tid & 7) << 3;          // 0..56  (B staging n)
  const _Float16* Bsrc = B + (long)bkk * ldb + bn + bnn;

  // A staging: 2 slots/thread
  const int ar[2] = { (tid + 0)   >> 2, (tid + 256) >> 2 };
  const int ac[2] = { ((tid + 0) & 3) << 3, ((tid + 256) & 3) << 3 };

  v8f acc[2][2];
  for (int i = 0; i < 2; ++i)
    for (int j = 0; j < 2; ++j)
      for (int e = 0; e < 8; ++e) acc[i][j][e] = 0.f;

  auto stageA = [&](int k0, int buf) {
    if (fullM) {
      #pragma unroll
      for (int it = 0; it < 2; ++it) {
        unsigned lds_off = (unsigned)(size_t)(&As[buf][ar[it]][ac[it]]);
        const _Float16* src = A + (long)(bm + ar[it]) * lda + k0 + ac[it];
        asm volatile("global_load_async_to_lds_b128 %0, %1, off"
                     :: "v"(lds_off), "v"(src) : "memory");
      }
    } else {
      #pragma unroll
      for (int it = 0; it < 2; ++it) {
        unsigned lds_off = (unsigned)(size_t)(&As[buf][ar[it]][ac[it]]);
        const _Float16* src = A + (long)(bm + ar[it]) * lda + k0 + ac[it];
        if (bm + ar[it] < M) {
          asm volatile("global_load_async_to_lds_b128 %0, %1, off"
                       :: "v"(lds_off), "v"(src) : "memory");
        } else {
          v8h z; for (int e = 0; e < 8; ++e) z[e] = (_Float16)0.f;
          *(v8h*)(&As[buf][ar[it]][ac[it]]) = z;
        }
      }
    }
  };
  auto storeB = [&](v8h bv, int buf) {
    #pragma unroll
    for (int e = 0; e < 8; ++e) Bs[buf][bnn + e][bkk] = bv[e];
  };

  // ---- prologue: stage tile 0 ----
  v8h bv = *(const v8h*)(Bsrc);
  stageA(0, 0);
  storeB(bv, 0);
  asm volatile("s_wait_asynccnt 0x0" ::: "memory");
  __syncthreads();

  const int nk = K >> 5;
  for (int t = 0; t < nk; ++t) {
    const int cur = t & 1, nxt = cur ^ 1;
    const bool more = (t + 1 < nk);
    if (more) {                               // issue next-tile loads early
      bv = *(const v8h*)(Bsrc + (long)(t + 1) * 32);
      stageA((t + 1) * 32, nxt);
    }

    // ---- fragments (A 16x32 / B 32x16, contiguous k = 16*hl + s in both) ----
    v16h af[2], bf[2];
    #pragma unroll
    for (int i = 0; i < 2; ++i) {
      int m = wm + i * 16 + l16;
      #pragma unroll
      for (int v = 0; v < 8; ++v) {
        int kk = 16 * hl + 2 * v;
        af[i][2 * v]     = As[cur][m][kk];
        af[i][2 * v + 1] = As[cur][m][kk + 1];
      }
    }
    #pragma unroll
    for (int j = 0; j < 2; ++j) {
      int n = wn + j * 16 + l16;
      #pragma unroll
      for (int v = 0; v < 8; ++v) {
        int kk = 16 * hl + 2 * v;
        bf[j][2 * v]     = Bs[cur][n][kk];
        bf[j][2 * v + 1] = Bs[cur][n][kk + 1];
      }
    }
    #pragma unroll
    for (int i = 0; i < 2; ++i)
      #pragma unroll
      for (int j = 0; j < 2; ++j)
        acc[i][j] = __builtin_amdgcn_wmma_f32_16x16x32_f16(
            false, af[i], false, bf[j], (short)0, acc[i][j], false, false);

    if (more) {                               // finish next-tile staging
      storeB(bv, nxt);
      asm volatile("s_wait_asynccnt 0x0" ::: "memory");
    }
    __syncthreads();
  }

  // ---- epilogue ----
  #pragma unroll
  for (int i = 0; i < 2; ++i) {
    #pragma unroll
    for (int j = 0; j < 2; ++j) {
      int ncol = bn + wn + j * 16 + l16;
      if (ncol >= N) continue;
      float bvs = bias ? bias[ncol] : 0.f;
      #pragma unroll
      for (int r = 0; r < 8; ++r) {
        int mr = bm + wm + i * 16 + r + 8 * hl;
        if (mr >= M) continue;
        float val = acc[i][j][r] + bvs;
        if (Cf) Cf[(long)mr * ldcf + ncol] = val;
        if (Ch) Ch[(long)mr * ldch + ncol] = (_Float16)val;
      }
    }
  }
}

// ---------------------------------------------------------------------------
// Pack K (B,H,SP,DH) and V^T (B,H,DH,SP) f16 buffers
// ---------------------------------------------------------------------------
__global__ void k_pack_kv(
    const _Float16* __restrict__ kst, const _Float16* __restrict__ vst,
    const float* __restrict__ mem_k, const float* __restrict__ mem_v,
    const _Float16* __restrict__ kt,  const _Float16* __restrict__ vt_tok,
    _Float16* __restrict__ Kb, _Float16* __restrict__ Vt)
{
  long idx = (long)blockIdx.x * blockDim.x + threadIdx.x;
  long total = (long)kB * kH * kSP * kDH;
  if (idx >= total) return;
  int  d  = (int)(idx % kDH);
  long t2 = idx / kDH;
  int  s  = (int)(t2 % kSP);
  long t3 = t2 / kSP;
  int  h  = (int)(t3 % kH);
  int  b  = (int)(t3 / kH);

  _Float16 kv = (_Float16)0.f, vv = (_Float16)0.f;
  if (s == 0) {
    kv = kst[b * kD + h * kDH + d];
    vv = vst[b * kD + h * kDH + d];
  } else if (s <= kM) {
    kv = (_Float16)mem_k[(h * kM + (s - 1)) * kDH + d];
    vv = (_Float16)mem_v[(h * kM + (s - 1)) * kDH + d];
  } else if (s < 1 + kM + kT) {
    int t = s - 1 - kM;
    kv = kt[(long)(b * kT + t) * kD + h * kDH + d];
    vv = vt_tok[(long)(b * kT + t) * kD + h * kDH + d];
  }
  Kb[((long)(b * kH + h) * kSP + s) * kDH + d] = kv;
  Vt[((long)(b * kH + h) * kDH + d) * kSP + s] = vv;
}

// ---------------------------------------------------------------------------
// Attention: one block = (b, h, 16-query tile). WMMA scores + softmax + WMMA ctx.
// ---------------------------------------------------------------------------
__global__ __launch_bounds__(256) void k_attn(
    const _Float16* __restrict__ q,
    const _Float16* __restrict__ Kb,
    const _Float16* __restrict__ Vt,
    float* __restrict__ ctx)
{
  __shared__ float    sc[16][kSP];
  __shared__ _Float16 att[16][kSP];
  __shared__ float    red[16][16];
  __shared__ float    rowmax[16], rowsum[16];

  const int tid  = threadIdx.x;
  const int wave = tid >> 5;
  const int lane = tid & 31;
  const int hl   = lane >> 4;
  const int l16  = lane & 15;
  const int nqt  = kT / 16;                       // 32
  const int qt   = blockIdx.x % nqt;
  const int h    = (blockIdx.x / nqt) % kH;
  const int b    = blockIdx.x / (nqt * kH);
  const float scale = 0.10206207261596575f;       // 1/sqrt(96)

  const _Float16* qbase = q  + (long)(b * kT + qt * 16) * kD + h * kDH;
  const _Float16* kbase = Kb + (long)(b * kH + h) * kSP * kDH;
  const _Float16* vbase = Vt + (long)(b * kH + h) * kDH * kSP;

  // ---- scores: q (16xDH) @ K^T (DHxS) ----
  for (int st = wave; st < kSP / 16; st += 8) {
    v8f c; for (int e = 0; e < 8; ++e) c[e] = 0.f;
    for (int kc = 0; kc < kDH; kc += 32) {
      v16h a, bb;
      const _Float16* qp = qbase + (long)l16 * kD + kc;
      #pragma unroll
      for (int v = 0; v < 8; ++v)
        ld_pair(qp + 16 * hl + 2 * v, a, v);
      const _Float16* kp = kbase + (long)(st * 16 + l16) * kDH + kc;
      #pragma unroll
      for (int v = 0; v < 8; ++v)
        ld_pair(kp + 16 * hl + 2 * v, bb, v);
      c = __builtin_amdgcn_wmma_f32_16x16x32_f16(false, a, false, bb, (short)0, c, false, false);
    }
    int s = st * 16 + l16;
    #pragma unroll
    for (int r = 0; r < 8; ++r) {
      float v = c[r] * scale;
      if (s >= kSreal) v = -1e30f;
      sc[r + 8 * hl][s] = v;
    }
  }
  __syncthreads();

  // ---- softmax over s ----
  {
    int r = tid >> 4, cth = tid & 15;
    float mx = -1e30f;
    for (int s = cth; s < kSP; s += 16) mx = fmaxf(mx, sc[r][s]);
    red[r][cth] = mx;
    __syncthreads();
    if (cth == 0) {
      float m = red[r][0];
      for (int i = 1; i < 16; ++i) m = fmaxf(m, red[r][i]);
      rowmax[r] = m;
    }
    __syncthreads();
    float m = rowmax[r], sum = 0.f;
    for (int s = cth; s < kSP; s += 16) {
      float e = __expf(sc[r][s] - m);
      att[r][s] = (_Float16)e;
      sum += e;
    }
    red[r][cth] = sum;
    __syncthreads();
    if (cth == 0) {
      float s0 = 0.f;
      for (int i = 0; i < 16; ++i) s0 += red[r][i];
      rowsum[r] = s0;
    }
    __syncthreads();
  }

  // ---- ctx: att (16xSP) @ V (SPxDH) ----
  if (wave < kDH / 16) {  // 6 waves
    v8f c; for (int e = 0; e < 8; ++e) c[e] = 0.f;
    const int dcol = wave * 16 + l16;
    const _Float16* vp0 = vbase + (long)dcol * kSP;
    for (int kc = 0; kc < kSP; kc += 32) {
      v16h a, bb;
      #pragma unroll
      for (int v = 0; v < 8; ++v)
        ld_pair(&att[l16][kc + 16 * hl + 2 * v], a, v);
      #pragma unroll
      for (int v = 0; v < 8; ++v)
        ld_pair(vp0 + kc + 16 * hl + 2 * v, bb, v);
      c = __builtin_amdgcn_wmma_f32_16x16x32_f16(false, a, false, bb, (short)0, c, false, false);
    }
    #pragma unroll
    for (int r = 0; r < 8; ++r) {
      int qr = r + 8 * hl;
      float val = c[r] / rowsum[qr];
      ctx[(long)(b * kT + qt * 16 + qr) * kD + h * kDH + dcol] = val;
    }
  }
}

// ---------------------------------------------------------------------------
// LayerNorm per row of ctx (f32 in), writes f16 for the W_o GEMM.
// ---------------------------------------------------------------------------
__global__ __launch_bounds__(128) void k_layernorm(
    const float* __restrict__ x, const float* __restrict__ g,
    const float* __restrict__ bta, _Float16* __restrict__ y)
{
  __shared__ float red[128];
  const int row = blockIdx.x, tid = threadIdx.x;
  const float* xr = x + (long)row * kD;
  float s = 0.f;
  for (int i = tid; i < kD; i += 128) s += xr[i];
  red[tid] = s; __syncthreads();
  for (int off = 64; off > 0; off >>= 1) { if (tid < off) red[tid] += red[tid + off]; __syncthreads(); }
  float mu = red[0] / kD;
  __syncthreads();
  float s2 = 0.f;
  for (int i = tid; i < kD; i += 128) { float d = xr[i] - mu; s2 += d * d; }
  red[tid] = s2; __syncthreads();
  for (int off = 64; off > 0; off >>= 1) { if (tid < off) red[tid] += red[tid + off]; __syncthreads(); }
  float rstd = rsqrtf(red[0] / kD + 1e-5f);
  for (int i = tid; i < kD; i += 128)
    y[(long)row * kD + i] = (_Float16)((xr[i] - mu) * rstd * g[i] + bta[i]);
}

// ---------------------------------------------------------------------------
// State update (tiny, f32 VALU)
// ---------------------------------------------------------------------------
__global__ void k_tanh_u(const float* __restrict__ state, const float* __restrict__ W_U,
                         float* __restrict__ t) {
  int idx = blockIdx.x * blockDim.x + threadIdx.x;
  if (idx >= kB * kRank) return;
  int b = idx / kRank, r = idx % kRank;
  float acc = 0.f;
  for (int k = 0; k < kDS; ++k) acc += state[b * kDS + k] * W_U[k * kRank + r];
  t[idx] = tanhf(acc);
}

__global__ void k_lowrank(const float* __restrict__ t, const float* __restrict__ W_V,
                          float* __restrict__ lr) {
  int idx = blockIdx.x * blockDim.x + threadIdx.x;
  if (idx >= kB * kDS) return;
  int b = idx / kDS, j = idx % kDS;
  float acc = 0.f;
  for (int r = 0; r < kRank; ++r) acc += t[b * kRank + r] * W_V[r * kDS + j];
  lr[idx] = acc;
}

__global__ void k_gate_state(const float* __restrict__ y, const float* __restrict__ state,
                             const float* __restrict__ W_g, const float* __restrict__ b_g,
                             const float* __restrict__ lr, float* __restrict__ out_state) {
  int idx = blockIdx.x * blockDim.x + threadIdx.x;
  if (idx >= kB * kDS) return;
  int b = idx / kDS, j = idx % kDS;
  float acc = b_g[j];
  const float* yrow = y + (long)(b * kT + kT - 1) * kD;   // last_ctx
  for (int i = 0; i < kD; ++i)  acc += yrow[i] * W_g[i * kDS + j];
  for (int i = 0; i < kDS; ++i) acc += state[b * kDS + i] * W_g[(kD + i) * kDS + j];
  float g = 1.f / (1.f + expf(-acc));
  out_state[idx] = state[idx] + g * lr[idx];
}

// ---------------------------------------------------------------------------
// Host launcher
// ---------------------------------------------------------------------------
extern "C" void kernel_launch(void* const* d_in, const int* in_sizes, int n_in,
                              void* d_out, int out_size, void* d_ws, size_t ws_size,
                              hipStream_t stream) {
  const int*   x_tok = (const int*)  d_in[0];
  const float* state = (const float*)d_in[1];
  const float* emb   = (const float*)d_in[2];
  const float* W_in  = (const float*)d_in[3];
  const float* b_in  = (const float*)d_in[4];
  const float* W_sk  = (const float*)d_in[5];
  const float* b_sk  = (const float*)d_in[6];
  const float* W_sv  = (const float*)d_in[7];
  const float* b_sv  = (const float*)d_in[8];
  const float* mem_k = (const float*)d_in[9];
  const float* mem_v = (const float*)d_in[10];
  const float* W_q   = (const float*)d_in[11];
  const float* b_q   = (const float*)d_in[12];
  const float* W_k   = (const float*)d_in[13];
  const float* b_k   = (const float*)d_in[14];
  const float* W_v   = (const float*)d_in[15];
  const float* b_v   = (const float*)d_in[16];
  const float* W_o   = (const float*)d_in[17];
  const float* b_o   = (const float*)d_in[18];
  const float* ln_g  = (const float*)d_in[19];
  const float* ln_b  = (const float*)d_in[20];
  const float* W_dec = (const float*)d_in[21];
  const float* W_U   = (const float*)d_in[22];
  const float* W_V   = (const float*)d_in[23];
  const float* W_g   = (const float*)d_in[24];
  const float* b_g   = (const float*)d_in[25];

  float* logits     = (float*)d_out;
  float* state_next = logits + (long)kB * kT * kVocab;

  char* wsp = (char*)d_ws;
  auto alloc = [&](size_t bytes) -> void* {
    void* p = (void*)wsp;
    wsp += (bytes + 255) & ~(size_t)255;
    return p;
  };
  _Float16* xe      = (_Float16*)alloc((size_t)kBT * kD * 2);
  _Float16* x_h     = (_Float16*)alloc((size_t)kBT * kD * 2);
  _Float16* q_h     = (_Float16*)alloc((size_t)kBT * kD * 2);
  _Float16* k_h     = (_Float16*)alloc((size_t)kBT * kD * 2);
  _Float16* v_h     = (_Float16*)alloc((size_t)kBT * kD * 2);
  _Float16* state_h = (_Float16*)alloc((size_t)kB * kDS * 2);
  _Float16* kst_h   = (_Float16*)alloc((size_t)kB * kD * 2);
  _Float16* vst_h   = (_Float16*)alloc((size_t)kB * kD * 2);
  _Float16* Kb      = (_Float16*)alloc((size_t)kB * kH * kSP * kDH * 2);
  _Float16* Vt      = (_Float16*)alloc((size_t)kB * kH * kDH * kSP * 2);
  float*    ctx     = (float*)   alloc((size_t)kBT * kD * 4);
  _Float16* yln_h   = (_Float16*)alloc((size_t)kBT * kD * 2);
  float*    y_f     = (float*)   alloc((size_t)kBT * kD * 4);
  _Float16* y_h     = (_Float16*)alloc((size_t)kBT * kD * 2);
  _Float16* W_in_h  = (_Float16*)alloc((size_t)kD * kD * 2);
  _Float16* W_q_h   = (_Float16*)alloc((size_t)kD * kD * 2);
  _Float16* W_k_h   = (_Float16*)alloc((size_t)kD * kD * 2);
  _Float16* W_v_h   = (_Float16*)alloc((size_t)kD * kD * 2);
  _Float16* W_o_h   = (_Float16*)alloc((size_t)kD * kD * 2);
  _Float16* W_sk_h  = (_Float16*)alloc((size_t)kDS * kD * 2);
  _Float16* W_sv_h  = (_Float16*)alloc((size_t)kDS * kD * 2);
  _Float16* W_dec_h = (_Float16*)alloc(((size_t)kD * kVocabPad + 256) * 2); // + slack
  float*    t_f     = (float*)   alloc((size_t)kB * kRank * 4);
  float*    lr_f    = (float*)   alloc((size_t)kB * kDS * 4);

  auto castf16 = [&](const float* s, _Float16* d, long n) {
    int blocks = (int)((n + 1023) / 1024); if (blocks > 16384) blocks = 16384;
    if (blocks < 1) blocks = 1;
    k_cast_f16<<<blocks, 256, 0, stream>>>(s, d, n);
  };
  auto gemm = [&](const _Float16* A, int lda, const _Float16* Bw, int ldb,
                  const float* bias, float* Cf, int ldcf, _Float16* Ch, int ldch,
                  int M, int N, int K) {
    dim3 g((N + 63) / 64, (M + 127) / 128);
    k_gemm_f16<<<g, 256, 0, stream>>>(A, lda, Bw, ldb, bias, Cf, ldcf, Ch, ldch, M, N, K);
  };

  // 1) weight / input casts to f16
  castf16(W_in, W_in_h, (long)kD * kD);
  castf16(W_q,  W_q_h,  (long)kD * kD);
  castf16(W_k,  W_k_h,  (long)kD * kD);
  castf16(W_v,  W_v_h,  (long)kD * kD);
  castf16(W_o,  W_o_h,  (long)kD * kD);
  castf16(W_sk, W_sk_h, (long)kDS * kD);
  castf16(W_sv, W_sv_h, (long)kDS * kD);
  castf16(state, state_h, (long)kB * kDS);
  k_cast_pad_f16<<<16384, 256, 0, stream>>>(W_dec, W_dec_h);

  // 2) embedding gather
  k_gather_emb<<<4096, 256, 0, stream>>>(x_tok, emb, xe);

  // 3) x = emb @ W_in + b_in
  gemm(xe, kD, W_in_h, kD, b_in, nullptr, 0, x_h, kD, kBT, kD, kD);

  // 4) q, k, v projections
  gemm(x_h, kD, W_q_h, kD, b_q, nullptr, 0, q_h, kD, kBT, kD, kD);
  gemm(x_h, kD, W_k_h, kD, b_k, nullptr, 0, k_h, kD, kBT, kD, kD);
  gemm(x_h, kD, W_v_h, kD, b_v, nullptr, 0, v_h, kD, kBT, kD, kD);

  // 5) state k/v projections
  gemm(state_h, kDS, W_sk_h, kD, b_sk, nullptr, 0, kst_h, kD, kB, kD, kDS);
  gemm(state_h, kDS, W_sv_h, kD, b_sv, nullptr, 0, vst_h, kD, kB, kD, kDS);

  // 6) pack K / V^T
  {
    long total = (long)kB * kH * kSP * kDH;
    int blocks = (int)((total + 255) / 256);
    k_pack_kv<<<blocks, 256, 0, stream>>>(kst_h, vst_h, mem_k, mem_v, k_h, v_h, Kb, Vt);
  }

  // 7) attention
  k_attn<<<kB * kH * (kT / 16), 256, 0, stream>>>(q_h, Kb, Vt, ctx);

  // 8) layernorm
  k_layernorm<<<kBT, 128, 0, stream>>>(ctx, ln_g, ln_b, yln_h);

  // 9) y = ln(ctx) @ W_o + b_o
  gemm(yln_h, kD, W_o_h, kD, b_o, y_f, kD, y_h, kD, kBT, kD, kD);

  // 10) logits = y @ W_dec  -> d_out (padded ldb, unpadded ldcf)
  gemm(y_h, kD, W_dec_h, kVocabPad, nullptr, logits, kVocab, nullptr, 0, kBT, kVocab, kD);

  // 11) state update
  k_tanh_u    <<<1, 256, 0, stream>>>(state, W_U, t_f);
  k_lowrank   <<<(kB * kDS + 255) / 256, 256, 0, stream>>>(t_f, W_V, lr_f);
  k_gate_state<<<(kB * kDS + 255) / 256, 256, 0, stream>>>(y_f, state, W_g, b_g, lr_f, state_next);

  (void)in_sizes; (void)n_in; (void)out_size; (void)ws_size;
}
